// MapEncoderPtsMA_27161373180462
// MI455X (gfx1250) — matrix operand order repaired and checked by reading, not compiled, not run on hardware
//
#include <hip/hip_runtime.h>
#include <hip/hip_bf16.h>
#include <math.h>

// ---------------------------------------------------------------------------
// MapEncoder fused for gfx1250 (wave32, WMMA f32<=f16 16x16x32).
// All six GEMMs (feats, scores, V, attn_out, FFN x2) run on the matrix pipe.
//
//   prep_qr    : R = wk @ q̂ per head (K-projection eliminated algebraically),
//                emitted as f16 B-layout (16x256, rows 8..15 zero) + c_h bias.
//   prep_wt    : f16 B-layout copies of wv, w1, w2, w_out, w_lin(K-padded).
//   attn_encode: per 8 rows (=160 tokens = 10 exact M-tiles, ~111 KB LDS --
//                gfx1250 WGs may allocate up to 320 KB):
//                feats(WMMA) -> scores(WMMA) -> masked softmax ->
//                V GEMM(WMMA) fused with attn-weighted ds_add_f32 scatter ->
//                attn_out(WMMA) -> LN1 -> emb f16.
//   ffn_ln     : per 32 rows (2 M-tiles share every B fetch):
//                emb@w1 relu (WMMA) -> @w2 (WMMA) with in-place residual
//                -> shuffle-reduced LN2 -> f32.
// ---------------------------------------------------------------------------

typedef _Float16 h16;
typedef __attribute__((ext_vector_type(16))) _Float16 v16h;
typedef __attribute__((ext_vector_type(8)))  _Float16 v8h;
typedef __attribute__((ext_vector_type(8)))  float    v8f;

union U16 { v16h v; v8h h[2]; };

#define D    256
#define FF   768
#define NH   8
#define HD   32
#define NP   20
#define KA   7
#define BNS  16384
#define EPS  1e-5f

#define RB   8            // rows per attn_encode workgroup
#define TOK  (RB * NP)    // 160 tokens
#define NMT  (TOK / 16)   // 10 M-tiles, exact

// workspace byte offsets (all 32B-aligned)
#define WS_EMB  0u                     // 16384*256 f16   (8,388,608 B)
#define WS_WVT  8388608u               // wv^T   [n][k] f16 (131,072 B)
#define WS_W1T  (WS_WVT + 131072u)     // w1^T   [f][d] f16 (393,216 B)
#define WS_W2T  (WS_W1T + 393216u)     // w2^T   [d][f] f16 (393,216 B)
#define WS_WOT  (WS_W2T + 393216u)     // w_out^T[d][k] f16 (131,072 B)
#define WS_WLT  (WS_WOT + 131072u)     // w_lin B-layout [d][32] f16 (16,384 B)
#define WS_RH   (WS_WLT + 16384u)      // R f16 [16][256] rows 8..15 zero (8,192 B)
#define WS_SB   (WS_RH + 8192u)        // 8 f32

__device__ __forceinline__ v8f wmma_f16(v16h a, v16h b, v8f c) {
  return __builtin_amdgcn_wmma_f32_16x16x32_f16(false, a, false, b,
                                                (short)0, c, false, false);
}

// ---------------------------------------------------------------------------
__global__ __launch_bounds__(256) void prep_qr(
    const float* __restrict__ map_seeds, const float* __restrict__ wq,
    const float* __restrict__ bq, const float* __restrict__ wk,
    const float* __restrict__ bk,
    h16* __restrict__ rh, float* __restrict__ sb) {
  __shared__ float s_qs[D];
  __shared__ float s_R[D * NH];
  const int tid = threadIdx.x;
  float s = bq[tid];
  for (int k = 0; k < D; ++k) s = fmaf(map_seeds[k], wq[k * D + tid], s);
  s *= 0.17677669529663687f;  // 1/sqrt(hd)
  s_qs[tid] = s;
  __syncthreads();
  for (int o = tid; o < D * NH; o += 256) {
    int k = o >> 3, h = o & 7;
    float acc = 0.f;
    for (int j = 0; j < HD; ++j)
      acc = fmaf(wk[k * D + h * HD + j], s_qs[h * HD + j], acc);
    s_R[o] = acc;
  }
  if (tid < NH) {
    float acc = 0.f;
    for (int j = 0; j < HD; ++j)
      acc = fmaf(bk[tid * HD + j], s_qs[tid * HD + j], acc);
    sb[tid] = acc;
  }
  __syncthreads();
  // B-layout f16: rh[n][k], n = head column (rows 8..15 zero-padded)
  for (int o = tid; o < 16 * D; o += 256) {
    int n = o >> 8, k = o & (D - 1);
    rh[o] = (h16)((n < NH) ? s_R[k * NH + n] : 0.f);
  }
}

// ---------------------------------------------------------------------------
__global__ __launch_bounds__(256) void prep_wt(
    const float* __restrict__ wv, const float* __restrict__ w1,
    const float* __restrict__ w2, const float* __restrict__ w_out,
    const float* __restrict__ w_lin,
    h16* __restrict__ wvt, h16* __restrict__ w1t, h16* __restrict__ w2t,
    h16* __restrict__ wot, h16* __restrict__ wlt) {
  const int idx0 = blockIdx.x * 256 + threadIdx.x;
  const int stride = gridDim.x * 256;
  for (int i = idx0; i < D * D; i += stride) {
    int n = i / D, k = i - n * D;
    wvt[n * D + k] = (h16)wv[k * D + n];
    wot[n * D + k] = (h16)w_out[k * D + n];
  }
  for (int i = idx0; i < D * FF; i += stride) {
    int f = i / D, d = i - f * D;
    w1t[f * D + d] = (h16)w1[d * FF + f];
  }
  for (int i = idx0; i < D * FF; i += stride) {
    int d = i / FF, f = i - d * FF;
    w2t[d * FF + f] = (h16)w2[f * D + d];
  }
  for (int i = idx0; i < D * 32; i += stride) {
    int n = i >> 5, k = i & 31;
    wlt[i] = (h16)((k < KA) ? w_lin[k * D + n] : 0.f);
  }
}

// ---------------------------------------------------------------------------
// 8 rows per workgroup -> 160 tokens -> exactly 10 WMMA M-tiles.
__global__ __launch_bounds__(256) void attn_encode(
    const float* __restrict__ roads, const float* __restrict__ b_lin,
    const float* __restrict__ bv, const float* __restrict__ b_out,
    const float* __restrict__ ln1_g, const float* __restrict__ ln1_b,
    const float* __restrict__ sbias, const h16* __restrict__ wvt,
    const h16* __restrict__ wot, const h16* __restrict__ wlt,
    const h16* __restrict__ rh, h16* __restrict__ emb_out) {
  __shared__ float s_attr[TOK][8];
  __shared__ int   s_mask[TOK];
  __shared__ int   s_allm[RB];
  __shared__ __align__(32) h16 s_feats[TOK * D];   // 80 KB
  __shared__ float s_sc[TOK][NH];
  __shared__ float s_ctx[RB * D];          // reused as attn_out after f16 copy
  __shared__ __align__(32) h16 s_cth[16 * D];
  __shared__ float s_p[RB][32], s_q[RB][32];
  __shared__ float s_mu[RB], s_rs[RB];

  const int tid  = threadIdx.x;
  const int lane = tid & 31;
  const int wave = tid >> 5;
  const int ng = lane >> 4;     // A: K-group, B: K-group, C: M-group
  const int ns = lane & 15;     // A: row m,   B/C: column n
  const int rowBase = blockIdx.x * RB;

  // --- attrs + padding mask (sum of attrs exactly 0) ---
  for (int t = tid; t < TOK; t += 256) {
    const float* rp = roads + ((size_t)rowBase * NP + t) * KA;
    float sum = 0.f;
    for (int a = 0; a < KA; ++a) { float x = rp[a]; s_attr[t][a] = x; sum += x; }
    s_mask[t] = (sum == 0.f) ? 1 : 0;
  }
  for (int o = tid; o < RB * D; o += 256) s_ctx[o] = 0.f;
  __syncthreads();
  if (tid < RB) {
    int all = 1;
    for (int p = 0; p < NP; ++p) all &= s_mask[tid * NP + p];
    s_allm[tid] = all;
  }

  // --- feats = attrs @ w_lin + b_lin via WMMA (K padded 7->32) ---
  for (int i = 0; i < 2; ++i) {
    const int nt = wave * 2 + i;
    const int dcol = nt * 16 + ns;
    const v16h bw = *(const v16h*)(wlt + dcol * 32 + ng * 16);
    const float bl = b_lin[dcol];
    for (int mt = 0; mt < NMT; ++mt) {
      U16 ua;
      #pragma unroll
      for (int e = 0; e < 16; ++e) ua.v[e] = (h16)0.f;
      if (ng == 0) {  // lanes 16..31 cover K=16..31 -> all zero pad
        const int tok = mt * 16 + ns;
        #pragma unroll
        for (int e = 0; e < KA; ++e) ua.v[e] = (h16)s_attr[tok][e];
      }
      v8f c = wmma_f16(ua.v, bw, (v8f){});
      #pragma unroll
      for (int r = 0; r < 8; ++r) {
        const int tok = mt * 16 + r + ng * 8;
        s_feats[tok * D + dcol] = (h16)(c[r] + bl);
      }
    }
  }
  __syncthreads();

  // --- scores = feats @ R^T + c via WMMA (N padded 8->16) ---
  for (int mt = wave; mt < NMT; mt += 8) {
    v8f acc = (v8f){};
    for (int ks = 0; ks < 8; ++ks) {
      v16h b = *(const v16h*)(rh + ns * D + ks * 32 + ng * 16);
      const int kA = ks * 32 + ng * 16;
      const int m = mt * 16 + ns;
      U16 ua;
      ua.h[0] = *(const v8h*)(&s_feats[m * D + kA]);
      ua.h[1] = *(const v8h*)(&s_feats[m * D + kA + 8]);
      acc = wmma_f16(ua.v, b, acc);
    }
    if (ns < NH) {
      const float sbh = sbias[ns];
      #pragma unroll
      for (int r = 0; r < 8; ++r) {
        const int tok = mt * 16 + r + ng * 8;
        s_sc[tok][ns] = acc[r] + sbh;
      }
    }
  }
  __syncthreads();

  // --- masked softmax per (row, head); fully-masked rows keep point 0 ---
  if (tid < RB * NH) {
    int row = tid >> 3, h = tid & 7;
    int allm = s_allm[row];
    float mx = -1e30f;
    for (int p = 0; p < NP; ++p) {
      int t = row * NP + p;
      bool m = s_mask[t] && !(allm && p == 0);
      if (!m) mx = fmaxf(mx, s_sc[t][h]);
    }
    float sum = 0.f;
    for (int p = 0; p < NP; ++p) {
      int t = row * NP + p;
      bool m = s_mask[t] && !(allm && p == 0);
      float w = m ? 0.f : __expf(s_sc[t][h] - mx);
      s_sc[t][h] = w; sum += w;
    }
    float inv = 1.f / sum;
    for (int p = 0; p < NP; ++p) s_sc[row * NP + p][h] *= inv;
  }
  __syncthreads();

  // --- V GEMM (WMMA) fused with attention-weighted scatter into ctx ---
  for (int i = 0; i < 2; ++i) {
    const int nt = wave * 2 + i;
    const int ncol = nt * 16 + ns;
    v8f acc[NMT];
    #pragma unroll
    for (int mt = 0; mt < NMT; ++mt) acc[mt] = (v8f){};
    for (int ks = 0; ks < 8; ++ks) {
      v16h b = *(const v16h*)(wvt + ncol * D + ks * 32 + ng * 16);
      const int kA = ks * 32 + ng * 16;
      #pragma unroll
      for (int mt = 0; mt < NMT; ++mt) {
        const int m = mt * 16 + ns;
        U16 ua;
        ua.h[0] = *(const v8h*)(&s_feats[m * D + kA]);
        ua.h[1] = *(const v8h*)(&s_feats[m * D + kA + 8]);
        acc[mt] = wmma_f16(ua.v, b, acc[mt]);
      }
    }
    const float bvn = bv[ncol];
    const int h = ncol >> 5;  // head of this output column
    #pragma unroll
    for (int mt = 0; mt < NMT; ++mt) {
      #pragma unroll
      for (int r = 0; r < 8; ++r) {
        const int tok = mt * 16 + r + ng * 8;   // 0..159, always valid
        const int row = tok / NP;
        const float w = s_sc[tok][h];
        atomicAdd(&s_ctx[row * D + ncol], (acc[mt][r] + bvn) * w);
      }
    }
  }
  __syncthreads();

  // --- ctx -> f16, M padded 8->16 ---
  for (int o = tid; o < 16 * D; o += 256) {
    int row = o >> 8;
    s_cth[o] = (h16)((row < RB) ? s_ctx[o] : 0.f);
  }
  __syncthreads();

  // --- attn_out = ctx @ w_out + b_out via WMMA (writes reuse s_ctx) ---
  float* s_ao = s_ctx;
  for (int i = 0; i < 2; ++i) {
    const int nt = wave * 2 + i;
    const int dcol = nt * 16 + ns;
    v8f acc = (v8f){};
    for (int ks = 0; ks < 8; ++ks) {
      v16h b = *(const v16h*)(wot + dcol * D + ks * 32 + ng * 16);
      const int kA = ks * 32 + ng * 16;
      U16 ua;
      ua.h[0] = *(const v8h*)(&s_cth[ns * D + kA]);
      ua.h[1] = *(const v8h*)(&s_cth[ns * D + kA + 8]);
      acc = wmma_f16(ua.v, b, acc);
    }
    const float bo = b_out[dcol];
    if (ng == 0) {  // M rows 0..7 (all real); ng=1 covers pad rows 8..15
      #pragma unroll
      for (int r = 0; r < 8; ++r) s_ao[r * D + dcol] = acc[r] + bo;
    }
  }
  __syncthreads();

  // --- LN1 -> emb (f16) ---
  {
    int row = tid >> 5, j = tid & 31;
    float ps = 0.f, pq = 0.f;
    #pragma unroll
    for (int u = 0; u < 8; ++u) {
      float x = s_ao[row * D + j * 8 + u];
      ps += x; pq += x * x;
    }
    s_p[row][j] = ps; s_q[row][j] = pq;
  }
  __syncthreads();
  if (tid < RB) {
    float ps = 0.f, pq = 0.f;
    for (int j = 0; j < 32; ++j) { ps += s_p[tid][j]; pq += s_q[tid][j]; }
    float mu = ps * (1.f / D);
    float var = pq * (1.f / D) - mu * mu;
    s_mu[tid] = mu; s_rs[tid] = rsqrtf(var + EPS);
  }
  __syncthreads();
  for (int o = tid; o < RB * D; o += 256) {
    int row = o >> 8, d = o & (D - 1);
    float e = (s_ao[o] - s_mu[row]) * s_rs[row] * ln1_g[d] + ln1_b[d];
    emb_out[(size_t)(rowBase + row) * D + d] = (h16)e;
  }
}

// ---------------------------------------------------------------------------
// 32 rows per workgroup (2 M-tiles amortize every B fetch):
// emb@w1 relu (WMMA) -> @w2 (WMMA) + in-place residual -> shuffle LN2.
__global__ __launch_bounds__(256) void ffn_ln(
    const h16* __restrict__ emb, const float* __restrict__ b1,
    const float* __restrict__ b2, const float* __restrict__ ln2_g,
    const float* __restrict__ ln2_b, const h16* __restrict__ w1t,
    const h16* __restrict__ w2t, float* __restrict__ out) {
  __shared__ __align__(32) h16 s_emb[32 * D];   // 16 KB; becomes x after GEMM2
  __shared__ __align__(32) h16 s_hid[32 * FF];  // 48 KB

  const int tid  = threadIdx.x;
  const int lane = tid & 31;
  const int wave = tid >> 5;
  const int base = blockIdx.x * 32;
  const int ng = lane >> 4;
  const int ns = lane & 15;

  {  // load emb tile (32x256 f16)
    const uint32_t* src = (const uint32_t*)(emb + (size_t)base * D);
    uint32_t* dst = (uint32_t*)s_emb;
    for (int o = tid; o < 32 * D / 2; o += 256) dst[o] = src[o];
  }
  __syncthreads();

  // --- GEMM1: hidden = relu(emb @ w1 + b1); 48 N-tiles, 8 K-steps ---
  for (int i = 0; i < 6; ++i) {
    const int nt = wave * 6 + i;
    const int f = nt * 16 + ns;
    v8f acc0 = (v8f){}, acc1 = (v8f){};
    for (int ks = 0; ks < 8; ++ks) {
      v16h b = *(const v16h*)(w1t + (size_t)f * D + ks * 32 + ng * 16);
      const int kA = ks * 32 + ng * 16;
      U16 a0, a1;
      a0.h[0] = *(const v8h*)(&s_emb[ns * D + kA]);
      a0.h[1] = *(const v8h*)(&s_emb[ns * D + kA + 8]);
      a1.h[0] = *(const v8h*)(&s_emb[(16 + ns) * D + kA]);
      a1.h[1] = *(const v8h*)(&s_emb[(16 + ns) * D + kA + 8]);
      acc0 = wmma_f16(a0.v, b, acc0);
      acc1 = wmma_f16(a1.v, b, acc1);
    }
    const float bb = b1[f];
    #pragma unroll
    for (int r = 0; r < 8; ++r) {
      const int t = r + ng * 8;
      s_hid[t * FF + f]        = (h16)fmaxf(acc0[r] + bb, 0.f);
      s_hid[(16 + t) * FF + f] = (h16)fmaxf(acc1[r] + bb, 0.f);
    }
  }
  __syncthreads();

  // --- GEMM2: x = hidden@w2 + b2 + emb (in-place into s_emb, f16).
  //     Each element is read+written only by its owning thread. ---
  for (int i = 0; i < 2; ++i) {
    const int nt = wave * 2 + i;
    const int d = nt * 16 + ns;
    v8f acc0 = (v8f){}, acc1 = (v8f){};
    for (int ks = 0; ks < 24; ++ks) {
      v16h b = *(const v16h*)(w2t + (size_t)d * FF + ks * 32 + ng * 16);
      const int kA = ks * 32 + ng * 16;
      U16 a0, a1;
      a0.h[0] = *(const v8h*)(&s_hid[ns * FF + kA]);
      a0.h[1] = *(const v8h*)(&s_hid[ns * FF + kA + 8]);
      a1.h[0] = *(const v8h*)(&s_hid[(16 + ns) * FF + kA]);
      a1.h[1] = *(const v8h*)(&s_hid[(16 + ns) * FF + kA + 8]);
      acc0 = wmma_f16(a0.v, b, acc0);
      acc1 = wmma_f16(a1.v, b, acc1);
    }
    const float bb = b2[d];
    #pragma unroll
    for (int r = 0; r < 8; ++r) {
      const int t = r + ng * 8;
      s_emb[t * D + d] = (h16)(acc0[r] + bb + (float)s_emb[t * D + d]);
      s_emb[(16 + t) * D + d] =
          (h16)(acc1[r] + bb + (float)s_emb[(16 + t) * D + d]);
    }
  }
  __syncthreads();

  // --- LN2 via in-wave shuffle reduction; wave w owns rows 4w..4w+3 ---
  for (int rr = 0; rr < 4; ++rr) {
    const int row = wave * 4 + rr;
    const v8h xv = *(const v8h*)(&s_emb[row * D + lane * 8]);
    float x[8];
    float ps = 0.f, pq = 0.f;
    #pragma unroll
    for (int e = 0; e < 8; ++e) {
      x[e] = (float)xv[e];
      ps += x[e]; pq += x[e] * x[e];
    }
    #pragma unroll
    for (int off = 16; off > 0; off >>= 1) {
      ps += __shfl_xor(ps, off, 32);
      pq += __shfl_xor(pq, off, 32);
    }
    const float mu = ps * (1.f / D);
    const float rs = rsqrtf(pq * (1.f / D) - mu * mu + EPS);
    float y[8];
    #pragma unroll
    for (int e = 0; e < 8; ++e) {
      const int d = lane * 8 + e;
      y[e] = (x[e] - mu) * rs * ln2_g[d] + ln2_b[d];
    }
    float4* op = (float4*)(out + (size_t)(base + row) * D + lane * 8);
    op[0] = make_float4(y[0], y[1], y[2], y[3]);
    op[1] = make_float4(y[4], y[5], y[6], y[7]);
  }
}

// ---------------------------------------------------------------------------
extern "C" void kernel_launch(void* const* d_in, const int* in_sizes, int n_in,
                              void* d_out, int out_size, void* d_ws,
                              size_t ws_size, hipStream_t stream) {
  const float* roads     = (const float*)d_in[0];
  const float* map_seeds = (const float*)d_in[1];
  const float* w_lin     = (const float*)d_in[2];
  const float* b_lin     = (const float*)d_in[3];
  const float* wq        = (const float*)d_in[4];
  const float* bq        = (const float*)d_in[5];
  const float* wk        = (const float*)d_in[6];
  const float* bk        = (const float*)d_in[7];
  const float* wv        = (const float*)d_in[8];
  const float* bv        = (const float*)d_in[9];
  const float* w_out     = (const float*)d_in[10];
  const float* b_out     = (const float*)d_in[11];
  const float* ln1_g     = (const float*)d_in[12];
  const float* ln1_b     = (const float*)d_in[13];
  const float* w1        = (const float*)d_in[14];
  const float* b1        = (const float*)d_in[15];
  const float* w2        = (const float*)d_in[16];
  const float* b2        = (const float*)d_in[17];
  const float* ln2_g     = (const float*)d_in[18];
  const float* ln2_b     = (const float*)d_in[19];
  float* out = (float*)d_out;

  char* ws = (char*)d_ws;
  h16*   embw = (h16*)(ws + WS_EMB);
  h16*   wvt  = (h16*)(ws + WS_WVT);
  h16*   w1t  = (h16*)(ws + WS_W1T);
  h16*   w2t  = (h16*)(ws + WS_W2T);
  h16*   wot  = (h16*)(ws + WS_WOT);
  h16*   wlt  = (h16*)(ws + WS_WLT);
  h16*   rh   = (h16*)(ws + WS_RH);
  float* sb   = (float*)(ws + WS_SB);

  prep_qr<<<1, 256, 0, stream>>>(map_seeds, wq, bq, wk, bk, rh, sb);
  prep_wt<<<512, 256, 0, stream>>>(wv, w1, w2, w_out, w_lin,
                                   wvt, w1t, w2t, wot, wlt);
  attn_encode<<<BNS / RB, 256, 0, stream>>>(roads, b_lin, bv, b_out, ln1_g,
                                            ln1_b, sb, wvt, wot, wlt, rh, embw);
  ffn_ln<<<BNS / 32, 256, 0, stream>>>(embw, b1, b2, ln2_g, ln2_b, w1t, w2t,
                                       out);
}
